// MultiHeadAttentionLayer_57569741635851
// MI455X (gfx1250) — matrix operand order repaired and checked by reading, not compiled, not run on hardware
//
#include <hip/hip_runtime.h>

#define NN      100000   // nodes
#define INDIM   128
#define NHEAD   8
#define HDIM    16
#define ROWT    5        // row tiles per block
#define ROWS    (ROWT * 16)   // 80 rows per block; 100000/80 = 1250 exact

typedef __attribute__((ext_vector_type(16))) _Float16 v16h;
typedef __attribute__((ext_vector_type(8)))  _Float16 v8h;
typedef __attribute__((ext_vector_type(8)))  float    v8f;

// ---------------------------------------------------------------------------
// Pack W (f32 [K=128][N=128]) into f16 WMMA B-fragment layout:
//   P[((colTile*4 + kStep)*32 + lane)*16 + j] = W[k][col]
//   with k = kStep*32 + 8*(lane>>4) + j (+8 if j>=8),  col = colTile*16 + (lane&15)
// so the GEMM loads each per-lane B fragment as one contiguous 32B vector.
// 1024 threads total per matrix (grid.y selects WQ/WK/WV).
// ---------------------------------------------------------------------------
__global__ __launch_bounds__(256) void pack_w_kernel(
    const float* __restrict__ WQ, const float* __restrict__ WK,
    const float* __restrict__ WV,
    _Float16* __restrict__ pQ, _Float16* __restrict__ pK,
    _Float16* __restrict__ pV)
{
    const float* W = (blockIdx.y == 0) ? WQ : (blockIdx.y == 1) ? WK : WV;
    _Float16*    P = (blockIdx.y == 0) ? pQ : (blockIdx.y == 1) ? pK : pV;

    const int tid = blockIdx.x * blockDim.x + threadIdx.x;   // 0..1023
    if (tid >= 1024) return;
    const int c    = tid >> 7;         // column tile 0..7
    const int kk   = (tid >> 5) & 3;   // K step 0..3
    const int lane = tid & 31;
    const int nn   = lane & 15;
    const int hf   = lane >> 4;

    _Float16* dst = P + (size_t)tid * 16;
#pragma unroll
    for (int j = 0; j < 16; ++j) {
        const int k = kk * 32 + 8 * hf + j + (j >= 8 ? 8 : 0);
        dst[j] = (_Float16)W[(size_t)k * INDIM + c * 16 + nn];
    }
}

// ---------------------------------------------------------------------------
// Projection GEMM: out[80 x 128] = h @ W + b per block (grid.y selects Q/K/V).
// 8 waves; wave w owns columns [16w,16w+16), keeps its 4 B fragments resident
// in VGPRs, and runs 5 row tiles x 4 K-steps = 20 v_wmma_f32_16x16x32_f16,
// with A fragments fed from a 20 KB LDS tile of h (f16).
// Fragment layouts per ISA 7.12.2:
//   A 16x32: lane L: m=L&15, half=L>>4; slot j -> k = 8*half + j (+8 if j>=8)
//   B 32x16: lane L: n=L&15; slot j -> k = 8*half + j (+8 if j>=8)   (pre-packed)
//   C/D:     VGPR r: row = r + 8*half, col = L&15
// ---------------------------------------------------------------------------
__global__ __launch_bounds__(256) void proj_wmma_kernel(
    const float* __restrict__ h,
    const _Float16* __restrict__ pQ, const _Float16* __restrict__ pK,
    const _Float16* __restrict__ pV,
    const float* __restrict__ bQ, const float* __restrict__ bK,
    const float* __restrict__ bV,
    float* __restrict__ oQ, float* __restrict__ oK, float* __restrict__ oV)
{
    const _Float16* P  = (blockIdx.y == 0) ? pQ : (blockIdx.y == 1) ? pK : pV;
    const float* bias  = (blockIdx.y == 0) ? bQ : (blockIdx.y == 1) ? bK : bV;
    float*       out   = (blockIdx.y == 0) ? oQ : (blockIdx.y == 1) ? oK : oV;

    __shared__ _Float16 ah[ROWS * INDIM];        // 80x128 f16 = 20 KB
    const int row0 = blockIdx.x * ROWS;

    // Stage h tile: float4 global loads -> packed f16 LDS stores
    for (int idx = threadIdx.x; idx < ROWS * INDIM / 4; idx += 256) {
        const float4 v = ((const float4*)(h + (size_t)row0 * INDIM))[idx];
        _Float16* d = &ah[idx * 4];
        d[0] = (_Float16)v.x; d[1] = (_Float16)v.y;
        d[2] = (_Float16)v.z; d[3] = (_Float16)v.w;
    }
    __syncthreads();

    const int wave = threadIdx.x >> 5;
    const int lane = threadIdx.x & 31;
    const int col0 = wave * 16;
    const int nn   = lane & 15;
    const int hf   = lane >> 4;

    // Resident B fragments: one 32B contiguous vector load each
    v16h b[4];
#pragma unroll
    for (int kk = 0; kk < 4; ++kk)
        b[kk] = *(const v16h*)(P + (size_t)(((wave * 4 + kk) * 32 + lane) * 16));

    const float bv = bias[col0 + nn];

#pragma unroll
    for (int rt = 0; rt < ROWT; ++rt) {
        const int mrow = rt * 16 + nn;
        v8f acc = {};
#pragma unroll
        for (int kk = 0; kk < 4; ++kk) {
            v8h lo = *(const v8h*)&ah[mrow * INDIM + kk * 32 + 8 * hf];
            v8h hi = *(const v8h*)&ah[mrow * INDIM + kk * 32 + 16 + 8 * hf];
            v16h a;
#pragma unroll
            for (int j = 0; j < 8; ++j) { a[j] = lo[j]; a[8 + j] = hi[j]; }
            acc = __builtin_amdgcn_wmma_f32_16x16x32_f16(
                /*neg_a=*/false, a, /*neg_b=*/false, b[kk],
                /*c_mod=*/(short)0, acc, /*reuse_a=*/false, /*reuse_b=*/false);
        }
#pragma unroll
        for (int r = 0; r < 8; ++r)
            out[(size_t)(row0 + rt * 16 + r + 8 * hf) * INDIM + col0 + nn] =
                acc[r] + bv;
    }
}

// ---------------------------------------------------------------------------
// Edge kernel: one thread per (edge, head). Q/K/V (153.6 MB) are L2-resident
// (192 MB L2). Scatter via agent-scope f32 fadd -> global_atomic_add_f32.
// ---------------------------------------------------------------------------
__device__ __forceinline__ void atomAddF(float* p, float v) {
    __hip_atomic_fetch_add(p, v, __ATOMIC_RELAXED, __HIP_MEMORY_SCOPE_AGENT);
}

__global__ __launch_bounds__(256) void edge_kernel(
    const float* __restrict__ Q, const float* __restrict__ K,
    const float* __restrict__ V, const int* __restrict__ src,
    const int* __restrict__ dst, float* __restrict__ wV,
    float* __restrict__ z, int nEdges)
{
    const int t = blockIdx.x * blockDim.x + threadIdx.x;
    if (t >= nEdges * NHEAD) return;
    const int e  = t >> 3;
    const int hd = t & 7;

    const int s = src[e];
    const int d = dst[e];

    const float4* q = (const float4*)(Q + (size_t)d * INDIM + hd * HDIM);
    const float4* k = (const float4*)(K + (size_t)s * INDIM + hd * HDIM);

    float dot = 0.f;
#pragma unroll
    for (int i = 0; i < 4; ++i) {
        float4 a = q[i], b = k[i];
        dot += a.x * b.x + a.y * b.y + a.z * b.z + a.w * b.w;
    }
    float sc = __expf(fminf(fmaxf(dot * 0.25f, -5.f), 5.f));  // 1/sqrt(16)=0.25

    const float4* v = (const float4*)(V + (size_t)s * INDIM + hd * HDIM);
    float* o = wV + (size_t)d * INDIM + hd * HDIM;
#pragma unroll
    for (int i = 0; i < 4; ++i) {
        float4 vv = v[i];
        atomAddF(o + 4 * i + 0, sc * vv.x);
        atomAddF(o + 4 * i + 1, sc * vv.y);
        atomAddF(o + 4 * i + 2, sc * vv.z);
        atomAddF(o + 4 * i + 3, sc * vv.w);
    }
    atomAddF(z + (size_t)d * NHEAD + hd, sc);
}

// ---------------------------------------------------------------------------
// Normalize: out[n][h][d] /= z[n][h]   (in place on d_out)
// ---------------------------------------------------------------------------
__global__ __launch_bounds__(256) void norm_kernel(
    float* __restrict__ out, const float* __restrict__ z)
{
    const int t = blockIdx.x * blockDim.x + threadIdx.x;
    if (t < NN * INDIM) out[t] = out[t] / z[t >> 4];   // t>>4 == n*8 + head
}

// ---------------------------------------------------------------------------
extern "C" void kernel_launch(void* const* d_in, const int* in_sizes, int n_in,
                              void* d_out, int out_size, void* d_ws, size_t ws_size,
                              hipStream_t stream)
{
    const float* h  = (const float*)d_in[0];
    const float* WQ = (const float*)d_in[1];
    const float* bQ = (const float*)d_in[2];
    const float* WK = (const float*)d_in[3];
    const float* bK = (const float*)d_in[4];
    const float* WV = (const float*)d_in[5];
    const float* bV = (const float*)d_in[6];
    const int*  src = (const int*)d_in[7];
    const int*  dst = (const int*)d_in[8];
    const int nEdges = in_sizes[7];

    float* out = (float*)d_out;

    // Workspace layout: Q | K | V | z (f32), then packed f16 weights (32B-aligned)
    const size_t NM = (size_t)NN * INDIM;
    float* Q = (float*)d_ws;
    float* K = Q + NM;
    float* V = K + NM;
    float* z = V + NM;
    _Float16* pQ = (_Float16*)(z + (size_t)NN * NHEAD);
    _Float16* pK = pQ + 128 * 128;
    _Float16* pV = pK + 128 * 128;

    hipMemsetAsync(out, 0, sizeof(float) * NM, stream);
    hipMemsetAsync(z,   0, sizeof(float) * (size_t)NN * NHEAD, stream);

    dim3 blk(256);

    pack_w_kernel<<<dim3(4, 3), blk, 0, stream>>>(WQ, WK, WV, pQ, pK, pV);

    proj_wmma_kernel<<<dim3(NN / ROWS, 3), blk, 0, stream>>>(
        h, pQ, pK, pV, bQ, bK, bV, Q, K, V);

    const int tot = nEdges * NHEAD;
    edge_kernel<<<(tot + 255) / 256, blk, 0, stream>>>(Q, K, V, src, dst, out, z, nEdges);

    norm_kernel<<<((NN * INDIM) + 255) / 256, blk, 0, stream>>>(out, z);
}
